// AddAttention_63153199120803
// MI455X (gfx1250) — matrix operand order repaired and checked by reading, not compiled, not run on hardware
//
#include <hip/hip_runtime.h>
#include <hip/hip_bf16.h>

typedef __attribute__((ext_vector_type(16))) _Float16 v16h;
typedef __attribute__((ext_vector_type(8)))  float    v8f;

#define B_SZ   32
#define TV_SZ  2048
#define D_SZ   512      // DQ == DV == ATT_DIM
#define TCH    64       // timesteps per chunk (amortizes WkT L2 traffic)
#define NCH    (TV_SZ / TCH)   // 32 chunks
#define NEG_INF (-1e9f)

// Branch-free tanh: gfx1250 hardware V_TANH_F32 when available.
__device__ __forceinline__ float fast_tanh(float x) {
#if __has_builtin(__builtin_amdgcn_tanhf)
    return __builtin_amdgcn_tanhf(x);
#elif __has_builtin(__builtin_amdgcn_tanh_f32)
    return __builtin_amdgcn_tanh_f32(x);
#else
    float ax = __builtin_fabsf(x);
    float e  = __expf(-2.0f * ax);
    float r  = (1.0f - e) * __builtin_amdgcn_rcpf(1.0f + e);
    return __builtin_copysignf(r, x);
#endif
}

// ---------------- kernel 1: q = query @ Wq + bq  (tiny) ----------------
__global__ __launch_bounds__(256)
void qproj_kernel(const float* __restrict__ query, const float* __restrict__ Wq,
                  const float* __restrict__ bq, float* __restrict__ qout) {
    int b = blockIdx.x, tid = threadIdx.x;
    __shared__ float qrow[D_SZ];
    qrow[tid]       = query[b * D_SZ + tid];
    qrow[tid + 256] = query[b * D_SZ + tid + 256];
    __syncthreads();
    for (int d = tid; d < D_SZ; d += 256) {
        float acc = bq[d];
        #pragma unroll 8
        for (int k = 0; k < D_SZ; ++k) acc = fmaf(qrow[k], Wq[k * D_SZ + d], acc);
        qout[b * D_SZ + d] = acc;
    }
}

// ------------- kernel 2: WkT[n][k] = (f16) Wk[k][n]  (512 KB, one-time) -------------
__global__ __launch_bounds__(256)
void wkprep_kernel(const float* __restrict__ Wk, _Float16* __restrict__ wkT) {
    int idx = blockIdx.x * 256 + threadIdx.x;       // 512*512 elements
    int n = idx & (D_SZ - 1);
    int k = idx >> 9;
    wkT[n * D_SZ + k] = (_Float16)Wk[k * D_SZ + n];
}

// ---------------- kernel 3: fused projection + scores + partial softmax/context ----------------
__global__ __launch_bounds__(256)
void addattn_main(const float* __restrict__ value, const int* __restrict__ lens,
                  const float* __restrict__ qproj, const float* __restrict__ bk,
                  const float* __restrict__ scale, const _Float16* __restrict__ wkT,
                  float* __restrict__ m_ws, float* __restrict__ l_ws,
                  float* __restrict__ ctx_ws) {
    const int c = blockIdx.x;          // chunk index
    const int b = blockIdx.y;          // batch
    const int tid = threadIdx.x;
    const int t0 = c * TCH;
    const int len = lens[b];
    const int pi = b * NCH + c;

    __shared__ __align__(16) _Float16 a16[TCH * D_SZ]; // 64 KB value tile (f16), row-major [t][k]
    __shared__ float2 qs[D_SZ];                        // {q[b,n]+bk[n], scale[n]}  (4 KB)
    __shared__ float sc[TCH];                          // raw scores for this chunk
    __shared__ float psc[TCH];                         // exp weights

    if (t0 >= len) {                   // fully masked chunk: emit neutral partial
        if (tid == 0) { m_ws[pi] = NEG_INF; l_ws[pi] = 0.f; }
        ctx_ws[(size_t)pi * D_SZ + tid] = 0.f;
        ctx_ws[(size_t)pi * D_SZ + 256 + tid] = 0.f;
        return;
    }

    // ---- stage: (q+bk, scale) table, zero scores, value tile f32 -> f16 LDS ----
    qs[tid]       = make_float2(qproj[b * D_SZ + tid] + bk[tid], scale[tid]);
    qs[tid + 256] = make_float2(qproj[b * D_SZ + tid + 256] + bk[tid + 256], scale[tid + 256]);
    if (tid < TCH) sc[tid] = 0.f;

    const float4* vsrc = (const float4*)(value + (size_t)(b * TV_SZ + t0) * D_SZ);
    uint2* a16v = (uint2*)a16;
    #pragma unroll
    for (int j = 0; j < TCH * D_SZ / 4 / 256; ++j) {   // 32 float4 per thread
        int i4 = tid + j * 256;
        float4 f = vsrc[i4];
        union { _Float16 h[4]; uint2 u; } pk;
        pk.h[0] = (_Float16)f.x; pk.h[1] = (_Float16)f.y;
        pk.h[2] = (_Float16)f.z; pk.h[3] = (_Float16)f.w;
        a16v[i4] = pk.u;
    }
    __syncthreads();

    // ---- WMMA: D[n][t] = WkT_tile (A: 16 att x 32 dv) x value^T tile (B: 32 dv x 16 t) ----
    // Per wave: 2 t-tiles (one half-chunk) x 8 N-tiles. B fragments (16 k-steps x 2 tiles
    // x 8 words) are re-read from LDS on every nt pass;
    // A streams from L2 and is reused for both t-tiles.
    const int wave = tid >> 5;
    const int lane = tid & 31;
    const int lm   = lane & 15;
    const int half = lane >> 4;
    const int ttp  = wave & 1;         // which half of the 64 timesteps (t base = ttp*32)
    const int ntg  = wave >> 1;        // N-tile group: 4 groups x 8 tiles
    const unsigned int* aV  = (const unsigned int*)a16;   // value tile (LDS)
    const unsigned int* bWk = (const unsigned int*)wkT;   // WkT (global, L2-resident)

    float s0 = 0.f, s1 = 0.f;          // score partials for t = ttp*32 + lm (+16)

    for (int nt = ntg * 8; nt < ntg * 8 + 8; ++nt) {
        const int nA = nt * 16 + lm;                    // A-matrix row this lane supplies
        v8f acc0 = {}; v8f acc1 = {};

        for (int k0 = 0; k0 < D_SZ; k0 += 32) {
            // A fragment (16x32 f16, ISA table):
            //  lanes 0-15: V0..3 K=k0+0..7,  V4..7 K=k0+16..23
            //  lanes16-31: V0..3 K=k0+8..15, V4..7 K=k0+24..31
            union { v16h h; unsigned int w[8]; } Af;
            const int g0 = (k0 + half * 8) >> 1;
            const int g1 = (k0 + 16 + half * 8) >> 1;
            #pragma unroll
            for (int r = 0; r < 4; ++r) {
                Af.w[r]     = bWk[nA * (D_SZ / 2) + g0 + r];
                Af.w[4 + r] = bWk[nA * (D_SZ / 2) + g1 + r];
            }
            // B fragments (32x16 f16): lane column = timestep, K along a16 row (contiguous).
            union { v16h h; unsigned int w[8]; } Bf0, Bf1;
            #pragma unroll
            for (int r = 0; r < 8; ++r) {
                const int kb = (r < 4) ? (g0 + r) : (g1 + r - 4);
                Bf0.w[r] = aV[(ttp * 32 + lm) * (D_SZ / 2) + kb];        // t = ttp*32+lm
                Bf1.w[r] = aV[(ttp * 32 + 16 + lm) * (D_SZ / 2) + kb];   // t = ttp*32+16+lm
            }
            acc0 = __builtin_amdgcn_wmma_f32_16x16x32_f16(false, Af.h, false, Bf0.h,
                                                          (short)0, acc0, false, false);
            acc1 = __builtin_amdgcn_wmma_f32_16x16x32_f16(false, Af.h, false, Bf1.h,
                                                          (short)0, acc1, false, false);
        }

        // D layout: VGPR r -> att row n = nt*16 + half*8 + r; lane -> timestep.
        #pragma unroll
        for (int r = 0; r < 8; ++r) {
            const float2 qsv = qs[nt * 16 + half * 8 + r];
            s0 = fmaf(qsv.y, fast_tanh(qsv.x + acc0[r]), s0);
            s1 = fmaf(qsv.y, fast_tanh(qsv.x + acc1[r]), s1);
        }
    }
    // 8 lanes (4 nt-groups x 2 halves) contribute per timestep: merge via LDS atomics.
    atomicAdd(&sc[ttp * 32 + lm], s0);
    atomicAdd(&sc[ttp * 32 + 16 + lm], s1);
    __syncthreads();

    // ---- per-chunk softmax partial (wave 0; 2 timesteps per lane) ----
    if (tid < 32) {
        const int ga = t0 + tid, gb = t0 + 32 + tid;
        float sa = (ga < len) ? sc[tid] : NEG_INF;
        float sb = (gb < len) ? sc[32 + tid] : NEG_INF;
        float mm = fmaxf(sa, sb);
        #pragma unroll
        for (int off = 16; off >= 1; off >>= 1) mm = fmaxf(mm, __shfl_xor(mm, off));
        float pa = (ga < len) ? __expf(sa - mm) : 0.f;
        float pb = (gb < len) ? __expf(sb - mm) : 0.f;
        psc[tid] = pa; psc[32 + tid] = pb;
        float l = pa + pb;
        #pragma unroll
        for (int off = 16; off >= 1; off >>= 1) l += __shfl_xor(l, off);
        if (tid == 0) { m_ws[pi] = mm; l_ws[pi] = l; }
    }
    __syncthreads();

    // ---- partial context: sum_t p[t] * value[b,t,:]  (re-read hits WGP$/L2) ----
    const float* vrow = value + (size_t)(b * TV_SZ + t0) * D_SZ;
    for (int dv = tid; dv < D_SZ; dv += 256) {
        float acc = 0.f;
        #pragma unroll 4
        for (int t = 0; t < TCH; ++t) acc = fmaf(psc[t], vrow[t * D_SZ + dv], acc);
        ctx_ws[(size_t)pi * D_SZ + dv] = acc;
    }
}

// ---------------- kernel 4: log-sum-exp merge of chunk partials ----------------
__global__ __launch_bounds__(256)
void addattn_reduce(const float* __restrict__ m_ws, const float* __restrict__ l_ws,
                    const float* __restrict__ ctx_ws, float* __restrict__ out) {
    const int b = blockIdx.x, tid = threadIdx.x;
    __shared__ float wgt[NCH];
    __shared__ float Lsh;
    if (tid < 32) {                      // NCH == 32: one chunk per lane
        float m1 = m_ws[b * NCH + tid];
        float mm = m1;
        #pragma unroll
        for (int off = 16; off >= 1; off >>= 1) mm = fmaxf(mm, __shfl_xor(mm, off));
        float w1 = __expf(m1 - mm);
        float l = w1 * l_ws[b * NCH + tid];
        #pragma unroll
        for (int off = 16; off >= 1; off >>= 1) l += __shfl_xor(l, off);
        wgt[tid] = w1;
        if (tid == 0) Lsh = l;
    }
    __syncthreads();
    const float invL = 1.0f / Lsh;
    for (int dv = tid; dv < D_SZ; dv += 256) {
        float acc = 0.f;
        #pragma unroll 4
        for (int c = 0; c < NCH; ++c)
            acc = fmaf(wgt[c], ctx_ws[(size_t)(b * NCH + c) * D_SZ + dv], acc);
        out[b * D_SZ + dv] = acc * invL;
    }
}

extern "C" void kernel_launch(void* const* d_in, const int* in_sizes, int n_in,
                              void* d_out, int out_size, void* d_ws, size_t ws_size,
                              hipStream_t stream) {
    const float* query = (const float*)d_in[0];
    const float* value = (const float*)d_in[1];
    const int*   lens  = (const int*)  d_in[2];
    const float* Wq    = (const float*)d_in[3];
    const float* bq    = (const float*)d_in[4];
    const float* Wk    = (const float*)d_in[5];
    const float* bk    = (const float*)d_in[6];
    const float* scale = (const float*)d_in[7];
    float* out = (float*)d_out;

    // workspace layout (bytes):
    //  [0, 64K)        q projection  (32*512 f32)
    //  [64K, 64K+512K) WkT f16       (512*512)
    //  then m[32*32], l[32*32], ctx[32*32*512] f32   (~2.6 MB total)
    char* ws = (char*)d_ws;
    float*     q_ws   = (float*)ws;
    _Float16*  wkT    = (_Float16*)(ws + 65536);
    float*     m_ws   = (float*)(ws + 65536 + 524288);
    float*     l_ws   = m_ws + B_SZ * NCH;
    float*     ctx_ws = l_ws + B_SZ * NCH;

    qproj_kernel<<<B_SZ, 256, 0, stream>>>(query, Wq, bq, q_ws);
    wkprep_kernel<<<(D_SZ * D_SZ) / 256, 256, 0, stream>>>(Wk, wkT);
    addattn_main<<<dim3(NCH, B_SZ), 256, 0, stream>>>(value, lens, q_ws, bk, scale,
                                                      wkT, m_ws, l_ws, ctx_ws);
    addattn_reduce<<<B_SZ, 256, 0, stream>>>(m_ws, l_ws, ctx_ws, out);
}
